// AttVlad_77790447665616
// MI455X (gfx1250) — matrix operand order repaired and checked by reading, not compiled, not run on hardware
//
#include <hip/hip_runtime.h>
#include <hip/hip_bf16.h>
#include <stdint.h>

typedef __attribute__((ext_vector_type(16))) __bf16 v16bf;
typedef __attribute__((ext_vector_type(8)))  __bf16 v8bf;
typedef __attribute__((ext_vector_type(4)))  __bf16 v4bf;
typedef __attribute__((ext_vector_type(8)))  float  v8f;

#define N_    32
#define D_    128
#define S_    16384
#define K_    64
#define STILE 64

// Build a 16-element bf16 A fragment from two contiguous 8-element LDS chunks.
__device__ __forceinline__ v16bf frag16(const __bf16* lo, const __bf16* hi) {
    v8bf a = *(const v8bf*)lo;
    v8bf b = *(const v8bf*)hi;
    return __builtin_shufflevector(a, b, 0, 1, 2, 3, 4, 5, 6, 7,
                                         8, 9, 10, 11, 12, 13, 14, 15);
}

__global__ __launch_bounds__(256)
void attvlad_main(const float* __restrict__ x, const float* __restrict__ conv_w,
                  const float* __restrict__ conv_b,
                  float* __restrict__ vlad_ws, float* __restrict__ asum_ws,
                  int CH, int SCH)
{
    __shared__ __bf16 xnB1[64][128];    // [s][d]  : B for GEMM1 (logits)
    __shared__ __bf16 xnB2[128][64];    // [d][s]  : B for GEMM2 (vlad)
    __shared__ float  logitsL[64][64];  // [s][k]
    __shared__ __bf16 aL[64][64];       // [k][s]  : A for GEMM2
    __shared__ float  pmaxL[4][64];
    __shared__ float  psumL[4][64];
    __shared__ float  normP16[16][64];  // per-d-row sumsq partials
    __shared__ float  rnormL[64];

    const int t    = threadIdx.x;
    const int lane = t & 31;
    const int w    = t >> 5;          // wave id 0..7
    const int n    = blockIdx.x / CH;
    const int ch   = blockIdx.x % CH;
    const int sbeg = ch * SCH;

    const int kt   = w >> 1;          // k-tile 0..3 (shared by GEMM1/GEMM2)
    const int wpar = w & 1;           // which half of s-tiles / d-tiles
    const int l15  = lane & 15;
    const int hi16 = lane >> 4;       // lane group 0/1
    const int c0A  = hi16 * 8;        // A-frag / C-frag row offset {0,8}
    const int c0B  = hi16 * 16;       // B-frag K offset {0,16}

    // loader mapping: 4 contiguous s-columns, 16 d-rows per pass (8 passes)
    const int jb = (t & 15) * 4;      // column base
    const int dr = t >> 4;            // d-row within pass, 0..15

    // ---- Preload conv_w A fragments (bf16) and bias C fragment (f32) ----
    v16bf Aw[4];
    v8f   biasF;
    {
        const int mrow = kt * 16 + l15;
        const float* wp = conv_w + mrow * D_;
        #pragma unroll
        for (int db = 0; db < 4; ++db) {
            v16bf f;
            #pragma unroll
            for (int i = 0; i < 8; ++i) {
                f[i]     = (__bf16)wp[db * 32 + c0B + i];
                f[i + 8] = (__bf16)wp[db * 32 + c0B + 8 + i];
            }
            Aw[db] = f;
        }
        #pragma unroll
        for (int j = 0; j < 8; ++j) biasF[j] = conv_b[kt * 16 + c0A + j];
    }

    v8f acc[4];
    #pragma unroll
    for (int u = 0; u < 4; ++u)
        #pragma unroll
        for (int j = 0; j < 8; ++j) acc[u][j] = 0.0f;

    float asum_r = 0.0f;              // threads 0..63 own k = t

    const int iters = SCH / STILE;
    for (int it = 0; it < iters; ++it) {
        const int s0 = sbeg + it * STILE;
        const float* pxb = x + (size_t)n * D_ * S_ + s0 + jb;

        // ---- Phase A: wide fp32 loads (b128), column sum-of-squares ----
        float4 xv[8];
        float ss0 = 0.f, ss1 = 0.f, ss2 = 0.f, ss3 = 0.f;
        #pragma unroll
        for (int p = 0; p < 8; ++p) {
            const float4 v = *(const float4*)(pxb + (size_t)(p * 16 + dr) * S_);
            xv[p] = v;
            ss0 += v.x * v.x; ss1 += v.y * v.y;
            ss2 += v.z * v.z; ss3 += v.w * v.w;
        }
        if (it + 1 < iters) __builtin_prefetch(pxb + (size_t)dr * S_ + STILE, 0, 0);
        *(float4*)&normP16[dr][jb] = make_float4(ss0, ss1, ss2, ss3);
        __syncthreads();
        if (t < 64) {
            float tot = 0.f;
            #pragma unroll
            for (int p = 0; p < 16; ++p) tot += normP16[p][t];
            rnormL[t] = 1.0f / fmaxf(sqrtf(tot), 1e-12f);
        }
        __syncthreads();
        {
            const float4 rn4 = *(const float4*)&rnormL[jb];
            #pragma unroll
            for (int p = 0; p < 8; ++p) {
                const int d = p * 16 + dr;
                const __bf16 h0 = (__bf16)(xv[p].x * rn4.x);
                const __bf16 h1 = (__bf16)(xv[p].y * rn4.y);
                const __bf16 h2 = (__bf16)(xv[p].z * rn4.z);
                const __bf16 h3 = (__bf16)(xv[p].w * rn4.w);
                xnB1[jb + 0][d] = h0;
                xnB1[jb + 1][d] = h1;
                xnB1[jb + 2][d] = h2;
                xnB1[jb + 3][d] = h3;
                *(v4bf*)&xnB2[d][jb] = (v4bf){h0, h1, h2, h3};  // packed b64 store
            }
        }
        __syncthreads();

        // ---- Phase B: GEMM1  logits[k][s] = conv_w @ xn + bias ----
        #pragma unroll
        for (int sti = 0; sti < 2; ++sti) {
            const int st   = wpar * 2 + sti;
            const int scol = st * 16 + l15;
            v16bf b[4];
            #pragma unroll
            for (int db = 0; db < 4; ++db)
                b[db] = *(const v16bf*)&xnB1[scol][db * 32 + c0B];
            v8f c = biasF;
            #pragma unroll
            for (int db = 0; db < 4; ++db)
                c = __builtin_amdgcn_wmma_f32_16x16x32_bf16(
                        false, Aw[db], false, b[db], (short)0, c, false, false);
            *(v8f*)&logitsL[scol][kt * 16 + c0A] = c;
        }
        __syncthreads();

        // ---- Phase C: softmax over K=64 per column (4-way split) ----
        const int sC = t & 63;
        const int kq = t >> 6;
        {
            const float* lr = &logitsL[sC][kq * 16];
            float m = lr[0];
            #pragma unroll
            for (int i = 1; i < 16; ++i) m = fmaxf(m, lr[i]);
            pmaxL[kq][sC] = m;
        }
        __syncthreads();
        float e16[16];
        {
            float m = fmaxf(fmaxf(pmaxL[0][sC], pmaxL[1][sC]),
                            fmaxf(pmaxL[2][sC], pmaxL[3][sC]));
            const float* lr = &logitsL[sC][kq * 16];
            float s = 0.0f;
            #pragma unroll
            for (int i = 0; i < 16; ++i) {
                float e = __expf(lr[i] - m);
                e16[i] = e;
                s += e;
            }
            psumL[kq][sC] = s;
        }
        __syncthreads();
        {
            float inv = 1.0f / (psumL[0][sC] + psumL[1][sC] +
                                psumL[2][sC] + psumL[3][sC]);
            #pragma unroll
            for (int i = 0; i < 16; ++i)
                aL[kq * 16 + i][sC] = (__bf16)(e16[i] * inv);
        }
        __syncthreads();

        // ---- asum[k] partial (deterministic: thread t owns row k=t) ----
        if (t < 64) {
            float s = 0.0f;
            #pragma unroll
            for (int c8 = 0; c8 < 8; ++c8) {
                v8bf chunk = *(const v8bf*)&aL[t][c8 * 8];
                #pragma unroll
                for (int i = 0; i < 8; ++i) s += (float)chunk[i];
            }
            asum_r += s;
        }

        // ---- Phase D: GEMM2  vlad[k][d] += a @ xn^T ----
        #pragma unroll
        for (int sb = 0; sb < 2; ++sb) {
            const int mrow = kt * 16 + l15;
            const v16bf Af = frag16(&aL[mrow][sb * 32 + c0B],
                                    &aL[mrow][sb * 32 + c0B + 8]);
            v16bf Bf[4];
            #pragma unroll
            for (int u = 0; u < 4; ++u) {
                const int dcol = wpar * 64 + u * 16 + l15;
                Bf[u] = *(const v16bf*)&xnB2[dcol][sb * 32 + c0B];
            }
            #pragma unroll
            for (int u = 0; u < 4; ++u)
                acc[u] = __builtin_amdgcn_wmma_f32_16x16x32_bf16(
                             false, Af, false, Bf[u], (short)0, acc[u], false, false);
        }
        __syncthreads();   // protect LDS tiles before next iteration rewrites
    }

    // ---- Epilogue: write this chunk's partials (exclusive slices) ----
    float* outp = vlad_ws + (size_t)blockIdx.x * K_ * D_;
    #pragma unroll
    for (int u = 0; u < 4; ++u) {
        const int dcol = wpar * 64 + u * 16 + l15;
        #pragma unroll
        for (int jj = 0; jj < 8; ++jj) {
            const int krow = kt * 16 + c0A + jj;
            outp[krow * D_ + dcol] = acc[u][jj];
        }
    }
    if (t < 64) asum_ws[(size_t)blockIdx.x * K_ + t] = asum_r;
}

__global__ __launch_bounds__(128)
void attvlad_final(const float* __restrict__ vlad_ws, const float* __restrict__ asum_ws,
                   const float* __restrict__ centroids, const float* __restrict__ att_w,
                   const float* __restrict__ att_b, float* __restrict__ out, int CH)
{
    __shared__ float red[128];
    const int d = threadIdx.x;
    const int n = blockIdx.x >> 6;
    const int k = blockIdx.x & 63;

    const float cv = centroids[k * D_ + d];
    const float aw = att_w[d];

    // reduction 1: soft attention scalar  satt = centroids[k]·att_w + att_b
    red[d] = cv * aw;
    __syncthreads();
    #pragma unroll
    for (int off = 64; off > 0; off >>= 1) {
        if (d < off) red[d] += red[d + off];
        __syncthreads();
    }
    const float satt = red[0] + att_b[0];
    __syncthreads();

    // sum chunk partials in fixed order (deterministic)
    float v = 0.0f, asum = 0.0f;
    for (int c = 0; c < CH; ++c) {
        v    += vlad_ws[(((size_t)n * CH + c) * K_ + k) * D_ + d];
        asum += asum_ws[((size_t)n * CH + c) * K_ + k];
    }
    const float r = satt * (v - asum * cv);

    // reduction 2: intra-normalize over D
    red[d] = r * r;
    __syncthreads();
    #pragma unroll
    for (int off = 64; off > 0; off >>= 1) {
        if (d < off) red[d] += red[d + off];
        __syncthreads();
    }
    const float rn = 1.0f / fmaxf(sqrtf(red[0]), 1e-12f);
    out[((size_t)n * K_ + k) * D_ + d] = r * rn;
}

extern "C" void kernel_launch(void* const* d_in, const int* in_sizes, int n_in,
                              void* d_out, int out_size, void* d_ws, size_t ws_size,
                              hipStream_t stream) {
    const float* x         = (const float*)d_in[0];
    const float* conv_w    = (const float*)d_in[1];
    const float* conv_b    = (const float*)d_in[2];
    const float* centroids = (const float*)d_in[3];
    const float* att_w     = (const float*)d_in[4];
    const float* att_b     = (const float*)d_in[5];
    float* out = (float*)d_out;

    // Workspace per chunk: vlad partial (K*D f32) + asum partial (K f32) per image.
    const size_t per_ch = (size_t)N_ * K_ * D_ * sizeof(float)
                        + (size_t)N_ * K_ * sizeof(float);
    int CH = 8;
    while (CH > 1 && (size_t)CH * per_ch > ws_size) CH >>= 1;
    const int SCH = S_ / CH;

    float* vlad_ws = (float*)d_ws;
    float* asum_ws = vlad_ws + (size_t)N_ * CH * K_ * D_;

    attvlad_main<<<N_ * CH, 256, 0, stream>>>(x, conv_w, conv_b,
                                              vlad_ws, asum_ws, CH, SCH);
    attvlad_final<<<N_ * K_, 128, 0, stream>>>(vlad_ws, asum_ws, centroids,
                                               att_w, att_b, out, CH);
}
